// BasicAttention_27934467293384
// MI455X (gfx1250) — compile-verified, matrix operand order checked by reading, not run on hardware
//
#include <hip/hip_runtime.h>
#include <hip/hip_bf16.h>

// ---------------------------------------------------------------------------
// BasicAttention forward for MI455X (gfx1250, wave32, WMMA, async-to-LDS).
//   out0: lhs_emb [B,Q,D] f32     out1: l2 (softmax probs) [B,Q,K] f32
// Pipeline (when ws_size permits):
//   k0: ys  -> bf16 row-major copy in d_ws
//   k1: values -> bf16 TRANSPOSED copy vT[B][D][K] in d_ws
//   k2: attention: per-WG 16-row Q tile; score row-block in LDS (320KB WGP);
//       per-wave double-buffered global_load_async_to_lds_b128 staging;
//       fragments via ds_load_b128; v_wmma_f32_16x16x32_bf16 everywhere.
// ---------------------------------------------------------------------------

typedef __attribute__((ext_vector_type(16))) __bf16 v16bf;
typedef __attribute__((ext_vector_type(8)))  float  v8f;

union Frag  { v16bf v; unsigned int u[8]; };
union FragQ { v16bf v; uint4 q[2]; };

constexpr int Bc = 8;
constexpr int Qc = 2048;
constexpr int Kc = 2048;
constexpr int Dc = 1024;

__device__ __forceinline__ unsigned int f2bf_rne(float f) {
    unsigned int u = __float_as_uint(f);
    u += 0x7FFFu + ((u >> 16) & 1u);   // round-to-nearest-even
    return u >> 16;
}
__device__ __forceinline__ unsigned int pack_bf16(float lo, float hi) {
    return f2bf_rne(lo) | (f2bf_rne(hi) << 16);
}

// ---------------------------------------------------------------------------
// Conversion passes (one-time, memory-bound).
// ---------------------------------------------------------------------------
__global__ void __launch_bounds__(256)
convert_rowmajor_bf16(const float* __restrict__ src, unsigned int* __restrict__ dst,
                      long npairs)
{
    for (long i = (long)blockIdx.x * blockDim.x + threadIdx.x; i < npairs;
         i += (long)gridDim.x * blockDim.x) {
        dst[i] = pack_bf16(src[2 * i], src[2 * i + 1]);
    }
}

// vT[b][d][kpair] = pack(values[b][2kp][d], values[b][2kp+1][d])
__global__ void __launch_bounds__(256)
convert_transpose_bf16(const float* __restrict__ v, unsigned int* __restrict__ vT)
{
    const long n = (long)Bc * Dc * (Kc / 2);
    for (long i = (long)blockIdx.x * blockDim.x + threadIdx.x; i < n;
         i += (long)gridDim.x * blockDim.x) {
        const int  kp = (int)(i % (Kc / 2));
        const long t  = i / (Kc / 2);
        const int  d  = (int)(t % Dc);
        const int  b  = (int)(t / Dc);
        const float* s = v + ((long)b * Kc + 2 * kp) * Dc + d;
        vT[i] = pack_bf16(s[0], s[Dc]);
    }
}

// ---------------------------------------------------------------------------
// Async stage: 16 rows x 128 bf16 elems (256B/row) -> per-wave LDS buffer.
// Each of the 32 lanes issues 8x b128 async loads (tracked by ASYNCcnt).
// ---------------------------------------------------------------------------
__device__ __forceinline__ void stage16x128(const unsigned char* g_row0,
                                            unsigned rowbytes,
                                            unsigned lds_off, int lane)
{
#pragma unroll
    for (int i = 0; i < 8; ++i) {
        const unsigned lin = (unsigned)(i * 32 + lane) * 16u;
        const unsigned r   = lin >> 8;       // row within 16-row slab
        const unsigned cb  = lin & 255u;     // byte within 256B row
        const unsigned char* g = g_row0 + (size_t)r * rowbytes + cb;
        const unsigned l = lds_off + r * 256u + cb;
        asm volatile("global_load_async_to_lds_b128 %0, %1, off"
                     :: "v"(l), "v"(g) : "memory");
    }
}

// ---------------------------------------------------------------------------
// Main attention kernel (bf16 operand copies already in workspace).
// Dynamic LDS layout:
//   sS     16*2048 f32              131072
//   sP     16*2048 bf16 (u32)        65536
//   sX     16*1024 bf16 (u32)        32768
//   sStage 4 waves * 2 * 4KB         32768
//   sRed   128 f32 + 2*16 f32          640
constexpr unsigned SMEM_MAIN = 131072u + 65536u + 32768u + 32768u + 512u + 128u;

__global__ void __launch_bounds__(128)
basic_attention_kernel(const float* __restrict__ xs,
                       const unsigned char* __restrict__ ybf,   // [B][K][D] bf16
                       const unsigned char* __restrict__ vTbf,  // [B][D][K] bf16
                       float* __restrict__ out_emb,
                       float* __restrict__ out_l2)
{
    extern __shared__ unsigned char smem_raw[];
    float*         sS      = (float*)smem_raw;                    // [16][Kc]
    unsigned int*  sP      = (unsigned int*)(sS + 16 * Kc);       // [16][Kc/2]
    unsigned int*  sX      = sP + 16 * (Kc / 2);                  // [16][Dc/2]
    unsigned char* sStage  = (unsigned char*)(sX + 16 * (Dc / 2));
    float*         sRed    = (float*)(sStage + 32768);
    float*         sRowMax = sRed + 128;
    float*         sRowSum = sRowMax + 16;

    const int b    = blockIdx.y;
    const int q0   = blockIdx.x * 16;
    const int tid  = threadIdx.x;
    const int lane = tid & 31;
    const int wave = tid >> 5;
    const int hl   = lane >> 4;
    const int mn   = lane & 15;

    unsigned char* myStage   = sStage + wave * 8192;
    const unsigned myStageOff = (unsigned)(unsigned long long)(void*)myStage;

    // ---- stage X tile to LDS as packed bf16 --------------------------------
    const float* xb = xs + ((long)b * Qc + q0) * Dc;
    for (int i = tid; i < 16 * (Dc / 2); i += 128) {
        sX[i] = pack_bf16(xb[2 * i], xb[2 * i + 1]);
    }
    __syncthreads();

    const float scale = 0.03125f;   // 1/sqrt(1024)
    const uint4* sX4 = (const uint4*)sX;

    // ---- S phase: S = X * Y^T / sqrt(D) ------------------------------------
    {
        const unsigned char* yb = ybf + (size_t)b * Kc * Dc * 2;
        for (int t = 0; t < Kc / 64; ++t) {             // 32 k-tiles per wave
            const int k0 = (wave + 4 * t) * 16;
            const unsigned char* rowbase = yb + (size_t)k0 * (Dc * 2);
            stage16x128(rowbase, Dc * 2, myStageOff, lane);
            v8f acc = {};
            for (int c = 0; c < 8; ++c) {               // 128-d chunks
                if (c < 7) {
                    stage16x128(rowbase + (c + 1) * 256, Dc * 2,
                                myStageOff + (((c + 1) & 1) << 12), lane);
                    asm volatile("s_wait_asynccnt 0x8" ::: "memory");
                } else {
                    asm volatile("s_wait_asynccnt 0x0" ::: "memory");
                }
                const uint4* sB4 = (const uint4*)(myStage + ((c & 1) << 12));
#pragma unroll
                for (int dt = 0; dt < 4; ++dt) {        // 4 wmma / chunk
                    FragQ A, Bf;
                    const int ab = mn * 128 + c * 16 + dt * 4 + hl;
                    A.q[0] = sX4[ab];
                    A.q[1] = sX4[ab + 2];
                    const int bb = mn * 16 + dt * 4 + hl;
                    Bf.q[0] = sB4[bb];
                    Bf.q[1] = sB4[bb + 2];
                    acc = __builtin_amdgcn_wmma_f32_16x16x32_bf16(
                              false, A.v, false, Bf.v, (short)0, acc, false, false);
                }
            }
#pragma unroll
            for (int r = 0; r < 8; ++r) {
                const int m = r + 8 * hl;               // C/D: M = r + 8*half
                sS[m * Kc + k0 + mn] = acc[r] * scale;
            }
        }
    }
    __syncthreads();

    // ---- softmax over K (f32 in LDS); emit l2 + bf16 P ---------------------
    const int row  = tid >> 3;
    const int seg  = tid & 7;
    const int base = row * Kc + seg * 256;

    float mx = -3.402823466e38f;
    for (int i = 0; i < 256; ++i) {
        const float s = sS[base + i];
        mx = (s > mx) ? s : mx;
    }
    sRed[tid] = mx;
    __syncthreads();
    if (seg == 0) {
        float m2 = sRed[row * 8];
        for (int j = 1; j < 8; ++j) {
            const float s = sRed[row * 8 + j];
            m2 = (s > m2) ? s : m2;
        }
        sRowMax[row] = m2;
    }
    __syncthreads();
    const float rmax = sRowMax[row];

    float sum = 0.f;
    for (int i = 0; i < 256; ++i) sum += __expf(sS[base + i] - rmax);
    sRed[tid] = sum;
    __syncthreads();
    if (seg == 0) {
        float s2 = 0.f;
        for (int j = 0; j < 8; ++j) s2 += sRed[row * 8 + j];
        sRowSum[row] = 1.0f / s2;
    }
    __syncthreads();
    const float rinv = sRowSum[row];

    float* l2row = out_l2 + ((long)b * Qc + q0 + row) * Kc + seg * 256;
    for (int i = 0; i < 256; i += 2) {
        const float p0 = __expf(sS[base + i]     - rmax) * rinv;
        const float p1 = __expf(sS[base + i + 1] - rmax) * rinv;
        l2row[i]     = p0;
        l2row[i + 1] = p1;
        sP[(base + i) >> 1] = pack_bf16(p0, p1);
    }
    __syncthreads();

    // ---- PV phase: emb = P * V  (V staged from transposed bf16 copy) -------
    {
        const unsigned char* vb = vTbf + (size_t)b * Dc * Kc * 2;
        const uint4* sP4 = (const uint4*)sP;
        for (int t = 0; t < Dc / 64; ++t) {             // 16 d-tiles per wave
            const int dc0 = (wave + 4 * t) * 16;
            const unsigned char* rowbase = vb + (size_t)dc0 * (Kc * 2);
            stage16x128(rowbase, Kc * 2, myStageOff, lane);
            v8f acc = {};
            for (int c = 0; c < 16; ++c) {              // 128-k chunks
                if (c < 15) {
                    stage16x128(rowbase + (c + 1) * 256, Kc * 2,
                                myStageOff + (((c + 1) & 1) << 12), lane);
                    asm volatile("s_wait_asynccnt 0x8" ::: "memory");
                } else {
                    asm volatile("s_wait_asynccnt 0x0" ::: "memory");
                }
                const uint4* sB4 = (const uint4*)(myStage + ((c & 1) << 12));
#pragma unroll
                for (int dt = 0; dt < 4; ++dt) {
                    FragQ A, Bf;
                    const int ab = mn * 256 + c * 16 + dt * 4 + hl;
                    A.q[0] = sP4[ab];
                    A.q[1] = sP4[ab + 2];
                    const int bb = mn * 16 + dt * 4 + hl;
                    Bf.q[0] = sB4[bb];
                    Bf.q[1] = sB4[bb + 2];
                    acc = __builtin_amdgcn_wmma_f32_16x16x32_bf16(
                              false, A.v, false, Bf.v, (short)0, acc, false, false);
                }
            }
#pragma unroll
            for (int r = 0; r < 8; ++r) {
                const int m = r + 8 * hl;
                out_emb[((long)b * Qc + q0 + m) * Dc + dc0 + mn] = acc[r];
            }
        }
    }
}

// ---------------------------------------------------------------------------
// Self-contained fallback (round-1 kernel): used if ws_size is too small.
// ---------------------------------------------------------------------------
constexpr unsigned SMEM_FALLBACK = 131072u + 65536u + 32768u + 512u + 64u + 64u;

__global__ void __launch_bounds__(128)
attention_fallback_kernel(const float* __restrict__ xs,
                          const float* __restrict__ ys,
                          const float* __restrict__ values,
                          float* __restrict__ out_emb,
                          float* __restrict__ out_l2)
{
    extern __shared__ unsigned char smem_raw[];
    float*        sS      = (float*)smem_raw;
    unsigned int* sP      = (unsigned int*)(sS + 16 * Kc);
    unsigned int* sX      = sP + 16 * (Kc / 2);
    float*        sRed    = (float*)(sX + 16 * (Dc / 2));
    float*        sRowMax = sRed + 128;
    float*        sRowSum = sRowMax + 16;

    const int b    = blockIdx.y;
    const int q0   = blockIdx.x * 16;
    const int tid  = threadIdx.x;
    const int lane = tid & 31;
    const int wave = tid >> 5;
    const int hl   = lane >> 4;
    const int mn   = lane & 15;

    const float* xb = xs + ((long)b * Qc + q0) * Dc;
    for (int i = tid; i < 16 * (Dc / 2); i += 128) {
        sX[i] = pack_bf16(xb[2 * i], xb[2 * i + 1]);
    }
    __syncthreads();

    const float scale = 0.03125f;
    const float* yb = ys + (long)b * Kc * Dc;
    for (int t = 0; t < Kc / 64; ++t) {
        const int k0 = (wave + 4 * t) * 16;
        v8f acc = {};
        for (int dt = 0; dt < Dc / 32; ++dt) {
            const int d0 = dt * 32;
            Frag A, Bf;
#pragma unroll
            for (int v = 0; v < 8; ++v) {
                const int kk = (v < 4) ? (hl * 8 + 2 * (v & 3))
                                       : (16 + hl * 8 + 2 * (v & 3));
                A.u[v] = sX[(mn * Dc + d0 + kk) >> 1];
                const float* yp = yb + (long)(k0 + mn) * Dc + d0 + kk;
                Bf.u[v] = pack_bf16(yp[0], yp[1]);
            }
            acc = __builtin_amdgcn_wmma_f32_16x16x32_bf16(
                      false, A.v, false, Bf.v, (short)0, acc, false, false);
        }
#pragma unroll
        for (int r = 0; r < 8; ++r) {
            sS[(r + 8 * hl) * Kc + k0 + mn] = acc[r] * scale;
        }
    }
    __syncthreads();

    const int row  = tid >> 3;
    const int seg  = tid & 7;
    const int base = row * Kc + seg * 256;

    float mx = -3.402823466e38f;
    for (int i = 0; i < 256; ++i) { const float s = sS[base + i]; mx = (s > mx) ? s : mx; }
    sRed[tid] = mx;
    __syncthreads();
    if (seg == 0) {
        float m2 = sRed[row * 8];
        for (int j = 1; j < 8; ++j) { const float s = sRed[row * 8 + j]; m2 = (s > m2) ? s : m2; }
        sRowMax[row] = m2;
    }
    __syncthreads();
    const float rmax = sRowMax[row];

    float sum = 0.f;
    for (int i = 0; i < 256; ++i) sum += __expf(sS[base + i] - rmax);
    sRed[tid] = sum;
    __syncthreads();
    if (seg == 0) {
        float s2 = 0.f;
        for (int j = 0; j < 8; ++j) s2 += sRed[row * 8 + j];
        sRowSum[row] = 1.0f / s2;
    }
    __syncthreads();
    const float rinv = sRowSum[row];

    float* l2row = out_l2 + ((long)b * Qc + q0 + row) * Kc + seg * 256;
    for (int i = 0; i < 256; i += 2) {
        const float p0 = __expf(sS[base + i]     - rmax) * rinv;
        const float p1 = __expf(sS[base + i + 1] - rmax) * rinv;
        l2row[i] = p0; l2row[i + 1] = p1;
        sP[(base + i) >> 1] = pack_bf16(p0, p1);
    }
    __syncthreads();

    const float* vb = values + (long)b * Kc * Dc;
    for (int t = 0; t < Dc / 64; ++t) {
        const int dc0 = (wave + 4 * t) * 16;
        v8f acc = {};
        for (int kt = 0; kt < Kc / 32; ++kt) {
            const int kb = kt * 32;
            Frag A, Bf;
#pragma unroll
            for (int v = 0; v < 8; ++v) {
                const int kk = (v < 4) ? (hl * 8 + 2 * (v & 3))
                                       : (16 + hl * 8 + 2 * (v & 3));
                A.u[v] = sP[(mn * Kc + kb + kk) >> 1];
                const long off = (long)(kb + kk) * Dc + dc0 + mn;
                Bf.u[v] = pack_bf16(vb[off], vb[off + Dc]);
            }
            acc = __builtin_amdgcn_wmma_f32_16x16x32_bf16(
                      false, A.v, false, Bf.v, (short)0, acc, false, false);
        }
#pragma unroll
        for (int r = 0; r < 8; ++r) {
            out_emb[((long)b * Qc + q0 + (r + 8 * hl)) * Dc + dc0 + mn] = acc[r];
        }
    }
}

// ---------------------------------------------------------------------------
extern "C" void kernel_launch(void* const* d_in, const int* in_sizes, int n_in,
                              void* d_out, int out_size, void* d_ws, size_t ws_size,
                              hipStream_t stream) {
    (void)in_sizes; (void)n_in; (void)out_size;
    const float* xs     = (const float*)d_in[0];
    const float* ys     = (const float*)d_in[1];
    // d_in[2] = mask_ys: dead code in reference, intentionally ignored.
    const float* values = (const float*)d_in[3];

    float* out_emb = (float*)d_out;
    float* out_l2  = (float*)d_out + (long)Bc * Qc * Dc;

    const size_t bf16_mat = (size_t)Bc * Kc * Dc * 2;   // 33,554,432 B each
    const dim3 grid(Qc / 16, Bc);
    const dim3 block(128);

    if (ws_size >= 2 * bf16_mat) {
        unsigned int* ybf  = (unsigned int*)d_ws;
        unsigned int* vTbf = (unsigned int*)((unsigned char*)d_ws + bf16_mat);

        const long npairs = (long)Bc * Kc * (Dc / 2);
        convert_rowmajor_bf16<<<2048, 256, 0, stream>>>(ys, ybf, npairs);
        convert_transpose_bf16<<<2048, 256, 0, stream>>>(values, vTbf);

        (void)hipFuncSetAttribute((const void*)basic_attention_kernel,
                                  hipFuncAttributeMaxDynamicSharedMemorySize,
                                  (int)SMEM_MAIN);
        basic_attention_kernel<<<grid, block, SMEM_MAIN, stream>>>(
            xs, (const unsigned char*)ybf, (const unsigned char*)vTbf,
            out_emb, out_l2);
    } else {
        (void)hipFuncSetAttribute((const void*)attention_fallback_kernel,
                                  hipFuncAttributeMaxDynamicSharedMemorySize,
                                  (int)SMEM_FALLBACK);
        attention_fallback_kernel<<<grid, block, SMEM_FALLBACK, stream>>>(
            xs, ys, values, out_emb, out_l2);
    }
}
